// antiRNNLayer_12292196401867
// MI455X (gfx1250) — matrix operand order repaired and checked by reading, not compiled, not run on hardware
//
#include <hip/hip_runtime.h>

#define UNITS   128
#define FT_DIM  64
#define TSTEPS  1024
#define BATCH   512
#define EPS_F   0.01f
#define GAMMA_F 0.01f

typedef __attribute__((ext_vector_type(16))) __bf16         v16bf;
typedef __attribute__((ext_vector_type(16))) unsigned short ushort16;
typedef __attribute__((ext_vector_type(8)))  float          v8f;

// float -> bf16, round-to-nearest-even
__device__ __forceinline__ unsigned short f2bf(float f) {
  unsigned int u = __builtin_bit_cast(unsigned int, f);
  u += 0x7FFFu + ((u >> 16) & 1u);
  return (unsigned short)(u >> 16);
}

// Inverse of the CDNA5 16-bit A-matrix (16x32) layout:
// element (row M, k) -> (lane, elem-index within the lane's 16 bf16)
__device__ __forceinline__ void amap(int M, int k, int& lane, int& i) {
  if (k < 8)       { lane = M;      i = k;      }
  else if (k < 16) { lane = M + 16; i = k - 8;  }
  else if (k < 24) { lane = M;      i = k - 8;  }
  else             { lane = M + 16; i = k - 16; }
}

__device__ __forceinline__ v8f wmma_bf16(v16bf a, v16bf b, v8f c) {
  // (neg_a, A, neg_b, B, c_mod, C, reuse_a, reuse_b)
  return __builtin_amdgcn_wmma_f32_16x16x32_bf16(false, a, false, b, (short)0, c,
                                                 false, false);
}

__device__ __forceinline__ float fast_tanh(float z) {
#if __has_builtin(__builtin_amdgcn_tanhf)
  return __builtin_amdgcn_tanhf(z);               // hardware V_TANH_F32
#else
  z = fminf(fmaxf(z, -20.0f), 20.0f);
  const float e = __builtin_amdgcn_exp2f(z * 2.8853900817779268f);  // e^{2z}
  return (e - 1.0f) * __builtin_amdgcn_rcpf(e + 1.0f);
#endif
}

__global__ __launch_bounds__(256) void anti_rnn_fused_kernel(
    const float* __restrict__ inputs, const float* __restrict__ V,
    const float* __restrict__ W, const float* __restrict__ bias,
    const float* __restrict__ x0, float* __restrict__ out) {
  // double-buffered bf16 A-fragment staging (lane-contiguous, 32B/lane/frag)
  __shared__ __align__(16) unsigned short s_u[2][2][512];  // u_t : 16x64
  __shared__ __align__(16) unsigned short s_x[2][4][512];  // x_t : 16x128

  const int tid  = threadIdx.x;
  const int lane = tid & 31;
  const int wave = tid >> 5;         // 8 waves: wave owns unit columns n0..n0+15
  const int n0   = wave * 16;
  const int b0   = blockIdx.x * 16;  // 16 batch rows per workgroup

  // ---- persistent B-fragments in registers: V (K=64) and M = W - W^T - g*I ----
  v16bf vB[2], mB[4];
  {
    const int N  = lane & 15;
    const int kb = (lane < 16) ? 0 : 8;
#pragma unroll
    for (int f = 0; f < 2; ++f) {
      ushort16 u;
#pragma unroll
      for (int i = 0; i < 16; ++i) {
        const int K = f * 32 + kb + (i < 8 ? i : i + 8);
        u[i] = f2bf(V[(size_t)K * UNITS + (n0 + N)]);
      }
      vB[f] = __builtin_bit_cast(v16bf, u);
    }
#pragma unroll
    for (int f = 0; f < 4; ++f) {
      ushort16 u;
#pragma unroll
      for (int i = 0; i < 16; ++i) {
        const int K = f * 32 + kb + (i < 8 ? i : i + 8);
        const int n = n0 + N;
        float m = W[(size_t)K * UNITS + n] - W[(size_t)n * UNITS + K];
        if (K == n) m -= GAMMA_F;
        u[i] = f2bf(m);
      }
      mB[f] = __builtin_bit_cast(v16bf, u);
    }
  }

  // bias as a persistent C-layout accumulator seed (no per-step splat movs)
  v8f biasvec;
  {
    const float bv = bias[n0 + (lane & 15)];
#pragma unroll
    for (int r = 0; r < 8; ++r) biasvec[r] = bv;
  }

  // ---- stage x0 (broadcast over batch rows) into s_x[0] ----
#pragma unroll
  for (int j = 0; j < 8; ++j) {
    const int idx  = tid * 8 + j;   // 0..2047
    const int frag = idx >> 9;
    const int rem  = idx & 511;
    const int L    = rem >> 4;
    const int i    = rem & 15;
    const int K = (L < 16) ? ((i < 8) ? i : i + 8) : ((i < 8) ? i + 8 : i + 16);
    s_x[0][frag][rem] = f2bf(x0[frag * 32 + K]);
  }

  // ---- precompute loop-invariant scatter offsets ----
  // u staging: this thread owns 4 consecutive f32 of u_t (row pm, feats pf..pf+3)
  const int pm = (tid * 4) >> 6;
  const int pf = (tid * 4) & 63;
  const float* usrc = inputs + (size_t)(b0 + pm) * TSTEPS * FT_DIM + pf;
  int uoff[4];
#pragma unroll
  for (int j = 0; j < 4; ++j) {
    const int k = pf + j;
    int L, i;
    amap(pm, k & 31, L, i);
    uoff[j] = (k >> 5) * 512 + L * 16 + i;   // flat index into s_u[buf]
  }

  // x scatter + output offsets: this lane's C-layout slot is (Mr=r+Mhi, n0+Nl)
  const int Nl  = lane & 15;
  const int Mhi = (lane >> 4) << 3;
  int xoff[8];
  size_t ooff[8];
#pragma unroll
  for (int r = 0; r < 8; ++r) {
    const int Mr = r + Mhi;
    const int c  = n0 + Nl;
    int L2, i2;
    amap(Mr, c & 31, L2, i2);
    xoff[r] = (c >> 5) * 512 + L2 * 16 + i2; // flat index into s_x[buf]
    ooff[r] = (size_t)(b0 + Mr) * UNITS + c;
  }

  // ---- stage u(t=0) ----
  {
    const float4 v = *reinterpret_cast<const float4*>(usrc);
    unsigned short* du = &s_u[0][0][0];
    du[uoff[0]] = f2bf(v.x); du[uoff[1]] = f2bf(v.y);
    du[uoff[2]] = f2bf(v.z); du[uoff[3]] = f2bf(v.w);
  }
  __syncthreads();

  // persistent f32 state in C/D layout
  float xs[8];
#pragma unroll
  for (int r = 0; r < 8; ++r) xs[r] = x0[n0 + Nl];

  float* outp = out;
  for (int t = 0; t < TSTEPS; ++t) {
    const int cur = t & 1;
    const int nxt = cur ^ 1;

    // load A-fragments (2x ds_load_b128 each)
    v16bf uA[2], xA[4];
#pragma unroll
    for (int f = 0; f < 2; ++f)
      uA[f] = *reinterpret_cast<const v16bf*>(&s_u[cur][f][lane * 16]);
#pragma unroll
    for (int f = 0; f < 4; ++f)
      xA[f] = *reinterpret_cast<const v16bf*>(&s_x[cur][f][lane * 16]);

    // overlap: stage u_{t+1} into the other buffer; prefetch u_{t+2}
    if (t + 1 < TSTEPS) {
      const float4 v =
          *reinterpret_cast<const float4*>(usrc + (size_t)(t + 1) * FT_DIM);
      unsigned short* du = &s_u[nxt][0][0];
      du[uoff[0]] = f2bf(v.x); du[uoff[1]] = f2bf(v.y);
      du[uoff[2]] = f2bf(v.z); du[uoff[3]] = f2bf(v.w);
    }
    if (t + 2 < TSTEPS)
      __builtin_prefetch(usrc + (size_t)(t + 2) * FT_DIM, 0, 1);

    // two independent accumulator chains for WMMA ILP; bias seeds chain U
    v8f accU = wmma_bf16(uA[0], vB[0], biasvec);   // u_t V  (K = 64)
    accU     = wmma_bf16(uA[1], vB[1], accU);
    v8f zero;
#pragma unroll
    for (int r = 0; r < 8; ++r) zero[r] = 0.0f;    // folds to inline C=0
    v8f accX = wmma_bf16(xA[0], mB[0], zero);      // x_t M  (K = 128)
    accX     = wmma_bf16(xA[1], mB[1], accX);
    accX     = wmma_bf16(xA[2], mB[2], accX);
    accX     = wmma_bf16(xA[3], mB[3], accX);

    unsigned short* dx = &s_x[nxt][0][0];
#pragma unroll
    for (int r = 0; r < 8; ++r) {
      const float z  = accU[r] + accX[r];
      const float xn = xs[r] + EPS_F * fast_tanh(z);
      xs[r] = xn;
      outp[ooff[r]]  = xn;          // states[t] = x_{t+1}
      dx[xoff[r]]    = f2bf(xn);    // next step's A-fragment
    }
    outp += (size_t)BATCH * UNITS;
    __syncthreads();  // single barrier per step (double-buffered staging)
  }
}

extern "C" void kernel_launch(void* const* d_in, const int* in_sizes, int n_in,
                              void* d_out, int out_size, void* d_ws,
                              size_t ws_size, hipStream_t stream) {
  const float* inputs = (const float*)d_in[0];
  const float* V      = (const float*)d_in[1];
  const float* W      = (const float*)d_in[2];
  const float* bias   = (const float*)d_in[3];
  const float* x0     = (const float*)d_in[4];
  float* out = (float*)d_out;
  (void)in_sizes; (void)n_in; (void)out_size; (void)d_ws; (void)ws_size;

  dim3 grid(BATCH / 16);   // 32 workgroups, each owns 16 batch rows for all T
  dim3 block(256);         // 8 wave32s: one 16-unit N-tile per wave
  hipLaunchKernelGGL(anti_rnn_fused_kernel, grid, block, 0, stream,
                     inputs, V, W, bias, x0, out);
}
